// FbacBlock_85495618995015
// MI455X (gfx1250) — matrix operand order repaired and checked by reading, not compiled, not run on hardware
//
// CDNA5 (gfx1250, MI455X) implementation of the point-cloud upsampler.
// Design:
//  * All dense layers run as bf16 WMMA GEMMs (v_wmma_f32_16x16x32_bf16),
//    one wave per 16xNT*16 C tile, weights pre-transposed/padded to [N,K] bf16.
//  * Dominant cost (8192x8192x32 feature-space kNN Gram) is computed with one
//    WMMA per 16x16 tile inside the kNN kernel; the point loop is unrolled x2
//    so two independent WMMA chains hide the 4-coexec WMMA->VALU RAW hazard,
//    and candidate sq-norms are fetched as one v8f vector load per tile.
//  * BatchNorm (training-mode batch stats) as GEMM -> column-stats -> fused
//    BN+act+bf16 downconvert.
//  * CrossTransformer: fusion = [pcd;pcd] => 2-NN of each point are its two
//    copies (pos_d == 0, fj == feat[i]); neighbor structure hardcoded, MLPs run.
#include <hip/hip_runtime.h>

typedef __attribute__((ext_vector_type(16))) __bf16 v16bf;
typedef __attribute__((ext_vector_type(8)))  __bf16 v8bf;
typedef __attribute__((ext_vector_type(8)))  float  v8f;

static inline int cdiv(int a, int b) { return (a + b - 1) / b; }

// ---------------------------------------------------------------------------
// Generic bf16 WMMA GEMM: C[M,N] = A[M,K](bf16) @ B[K,N](bf16, given as BT[N,K]) + bias
// One wave per block. M%16==0, K%32==0, N%(NT*16)==0.
// ---------------------------------------------------------------------------
template <int NT>
__global__ void __launch_bounds__(32)
gemm_bf16_wmma(const __bf16* __restrict__ A, const __bf16* __restrict__ BT,
               const float* __restrict__ bias, float* __restrict__ C,
               int M, int N, int K) {
  const int lane = threadIdx.x & 31;
  const int half = lane >> 4;       // 0: K-low half rows, 1: K-high half rows
  const int r    = lane & 15;
  const int m0   = blockIdx.x * 16;
  const int n0   = blockIdx.y * (NT * 16);

  v8f acc[NT];
#pragma unroll
  for (int t = 0; t < NT; ++t)
#pragma unroll
    for (int v = 0; v < 8; ++v) acc[t][v] = 0.0f;

  for (int k0 = 0; k0 < K; k0 += 32) {
    // A fragment (16x32 bf16): lane<16 holds K{0..15}, lane>=16 K{16..31}
    const __bf16* arow = A + (size_t)(m0 + r) * K + k0 + half * 16;
    v8bf lo = *(const v8bf*)(arow);
    v8bf hi = *(const v8bf*)(arow + 8);
    v16bf a;
#pragma unroll
    for (int i = 0; i < 8; ++i) { a[i] = lo[i]; a[i + 8] = hi[i]; }
#pragma unroll
    for (int t = 0; t < NT; ++t) {
      // B fragment (32x16): lane holds column n, 16 consecutive K at half*16.
      const int n = n0 + t * 16 + r;
      v16bf b = *(const v16bf*)(BT + (size_t)n * K + k0 + half * 16);
      acc[t] = __builtin_amdgcn_wmma_f32_16x16x32_bf16(
          false, a, false, b, (short)0, acc[t], false, false);
    }
  }
  // C layout: lane holds col n0+t*16+r, rows m0 + 8*half + v
#pragma unroll
  for (int t = 0; t < NT; ++t) {
    const int n  = n0 + t * 16 + r;
    const float bv = bias[n];
#pragma unroll
    for (int v = 0; v < 8; ++v)
      C[(size_t)(m0 + half * 8 + v) * N + n] = acc[t][v] + bv;
  }
}

// ---------------------------------------------------------------------------
// Weight prep: W[K,N] f32 -> WT[Np,Kp] bf16 (transposed, zero padded)
// ---------------------------------------------------------------------------
__global__ void prep_wT_kernel(const float* __restrict__ w, __bf16* __restrict__ wT,
                               int K, int Nn, int Kp, int Np) {
  int t = blockIdx.x * blockDim.x + threadIdx.x;
  if (t >= Np * Kp) return;
  int n = t / Kp, k = t % Kp;
  wT[t] = (k < K && n < Nn) ? (__bf16)w[(size_t)k * Nn + n] : (__bf16)0.0f;
}

__global__ void pad_bias16(const float* __restrict__ b, float* __restrict__ bp) {
  int t = threadIdx.x;
  if (t < 16) bp[t] = (t < 3) ? b[t] : 0.0f;
}

// ---------------------------------------------------------------------------
// BatchNorm (batch stats) pipeline
// ---------------------------------------------------------------------------
__global__ void zero_f32(float* p, int n) {
  int i = blockIdx.x * blockDim.x + threadIdx.x;
  if (i < n) p[i] = 0.0f;
}

__global__ void colstats_accum(const float* __restrict__ x, float* __restrict__ accum,
                               int M, int N) {
  // blockDim.x = 256, N in {32,64,128}
  const int c = threadIdx.x % N;
  const int g = threadIdx.x / N;
  const int G = blockDim.x / N;
  float s = 0.0f, ss = 0.0f;
  for (int m = blockIdx.x * G + g; m < M; m += gridDim.x * G) {
    float v = x[(size_t)m * N + c];
    s += v; ss += v * v;
  }
  __shared__ float lds[512];
  lds[threadIdx.x] = s;
  lds[256 + threadIdx.x] = ss;
  __syncthreads();
  if (g == 0) {
    for (int k = 1; k < G; ++k) { s += lds[k * N + c]; ss += lds[256 + k * N + c]; }
    atomicAdd(&accum[c], s);
    atomicAdd(&accum[N + c], ss);
  }
}

__global__ void finalize_stats(const float* __restrict__ accum, float* __restrict__ mr,
                               int M, int N) {
  int c = blockIdx.x * blockDim.x + threadIdx.x;
  if (c >= N) return;
  float mean = accum[c] / (float)M;
  float var  = accum[N + c] / (float)M - mean * mean;
  mr[c]     = mean;
  mr[N + c] = rsqrtf(var + 1e-5f);
}

__global__ void bn_act_bf16(const float* __restrict__ x, const float* __restrict__ mr,
                            const float* __restrict__ g, const float* __restrict__ be,
                            __bf16* __restrict__ y, int M, int N, int leaky) {
  int i = blockIdx.x * blockDim.x + threadIdx.x;
  if (i >= M * N) return;
  int c = i % N;
  float v = (x[i] - mr[c]) * mr[N + c] * g[c] + be[c];
  v = (v > 0.0f) ? v : (leaky ? 0.2f * v : 0.0f);
  y[i] = (__bf16)v;
}

// ---------------------------------------------------------------------------
// kNN on positions (k=16, 3-dim) — scalar with LDS point tiling.
// ---------------------------------------------------------------------------
__global__ void __launch_bounds__(256)
knn_pos_kernel(const float* __restrict__ pcd, int* __restrict__ idx, int N) {
  const int TILE = 256;
  __shared__ float sx[TILE], sy[TILE], sz[TILE];
  const int q = blockIdx.x * blockDim.x + threadIdx.x;
  const float qx = pcd[3 * q], qy = pcd[3 * q + 1], qz = pcd[3 * q + 2];
  float bd[16]; int bi[16];
#pragma unroll
  for (int s = 0; s < 16; ++s) { bd[s] = 3.4e38f; bi[s] = 0; }
  for (int t0 = 0; t0 < N; t0 += TILE) {
    __syncthreads();
    int j = threadIdx.x;
    sx[j] = pcd[3 * (t0 + j)];
    sy[j] = pcd[3 * (t0 + j) + 1];
    sz[j] = pcd[3 * (t0 + j) + 2];
    __syncthreads();
    for (int jj = 0; jj < TILE; ++jj) {
      float dx = qx - sx[jj], dy = qy - sy[jj], dz = qz - sz[jj];
      float d = dx * dx + dy * dy + dz * dz;
      if (d < bd[15]) {
        bd[15] = d; bi[15] = t0 + jj;
#pragma unroll
        for (int s = 15; s > 0; --s) {
          if (bd[s] < bd[s - 1]) {
            float td = bd[s]; bd[s] = bd[s - 1]; bd[s - 1] = td;
            int ti = bi[s]; bi[s] = bi[s - 1]; bi[s - 1] = ti;
          }
        }
      }
    }
  }
#pragma unroll
  for (int s = 0; s < 16; ++s) idx[(size_t)q * 16 + s] = bi[s];
}

// ---------------------------------------------------------------------------
// kNN in 32-dim feature space (k=8) — WMMA Gram tiles (K=32 = feature width).
// 4 waves/block; each wave owns a 16-query tile. Query n lives in lanes n and
// n+16 of the C tile (each sees half of every 16-point candidate tile); the
// two per-lane top-8 lists are merged through LDS at the end.
// Point loop unrolled x2: two independent WMMA chains hide the WMMA->VALU
// RAW hazard; candidate norms fetched as one 32B vector load per tile half.
// ---------------------------------------------------------------------------
__global__ void __launch_bounds__(128)
knn_feat_wmma(const __bf16* __restrict__ F, const float* __restrict__ nrm,
              int* __restrict__ idx, int N) {
  const int lane = threadIdx.x & 31;
  const int wave = threadIdx.x >> 5;
  const int half = lane >> 4, r = lane & 15;
  const int q0 = (blockIdx.x * 4 + wave) * 16;
  __shared__ float sD[128 * 8];
  __shared__ int   sI[128 * 8];

  // B fragment: query tile, loaded once (K=32 covers the whole feature dim)
  v16bf b = *(const v16bf*)(F + (size_t)(q0 + r) * 32 + half * 16);
  const float qn = nrm[q0 + r];

  float bd[8]; int bi[8];
#pragma unroll
  for (int s = 0; s < 8; ++s) { bd[s] = 3.4e38f; bi[s] = 0; }

  for (int p0 = 0; p0 < N; p0 += 32) {
    // Two candidate tiles per iteration.
    const __bf16* ar0 = F + (size_t)(p0 + r) * 32 + half * 16;
    const __bf16* ar1 = F + (size_t)(p0 + 16 + r) * 32 + half * 16;
    v8bf lo0 = *(const v8bf*)(ar0);
    v8bf hi0 = *(const v8bf*)(ar0 + 8);
    v8bf lo1 = *(const v8bf*)(ar1);
    v8bf hi1 = *(const v8bf*)(ar1 + 8);
    v16bf a0, a1;
#pragma unroll
    for (int i = 0; i < 8; ++i) {
      a0[i] = lo0[i]; a0[i + 8] = hi0[i];
      a1[i] = lo1[i]; a1[i + 8] = hi1[i];
    }
    // Candidate sq-norms: one 32B vector load per tile half (8 contiguous f32).
    v8f pn0 = *(const v8f*)(nrm + p0 + 8 * half);
    v8f pn1 = *(const v8f*)(nrm + p0 + 16 + 8 * half);

    v8f c0, c1;
#pragma unroll
    for (int v = 0; v < 8; ++v) { c0[v] = 0.0f; c1[v] = 0.0f; }
    c0 = __builtin_amdgcn_wmma_f32_16x16x32_bf16(false, a0, false, b, (short)0, c0,
                                                 false, false);
    c1 = __builtin_amdgcn_wmma_f32_16x16x32_bf16(false, a1, false, b, (short)0, c1,
                                                 false, false);
#pragma unroll
    for (int v = 0; v < 8; ++v) {
      const int p = p0 + 8 * half + v;
      const float d = qn + pn0[v] - 2.0f * c0[v];
      if (d < bd[7]) {
        bd[7] = d; bi[7] = p;
#pragma unroll
        for (int s = 7; s > 0; --s) {
          if (bd[s] < bd[s - 1]) {
            float td = bd[s]; bd[s] = bd[s - 1]; bd[s - 1] = td;
            int ti = bi[s]; bi[s] = bi[s - 1]; bi[s - 1] = ti;
          }
        }
      }
    }
#pragma unroll
    for (int v = 0; v < 8; ++v) {
      const int p = p0 + 16 + 8 * half + v;
      const float d = qn + pn1[v] - 2.0f * c1[v];
      if (d < bd[7]) {
        bd[7] = d; bi[7] = p;
#pragma unroll
        for (int s = 7; s > 0; --s) {
          if (bd[s] < bd[s - 1]) {
            float td = bd[s]; bd[s] = bd[s - 1]; bd[s - 1] = td;
            int ti = bi[s]; bi[s] = bi[s - 1]; bi[s - 1] = ti;
          }
        }
      }
    }
  }
#pragma unroll
  for (int s = 0; s < 8; ++s) {
    sD[threadIdx.x * 8 + s] = bd[s];
    sI[threadIdx.x * 8 + s] = bi[s];
  }
  __syncthreads();
  if (lane < 16) {
    const int b0 = (wave * 32 + lane) * 8;
    const int b1 = (wave * 32 + lane + 16) * 8;
    int a0 = 0, a1 = 0;
    for (int s = 0; s < 8; ++s) {
      float d0 = (a0 < 8) ? sD[b0 + a0] : 3.4e38f;
      float d1 = (a1 < 8) ? sD[b1 + a1] : 3.4e38f;
      int o;
      if (d0 <= d1) { o = sI[b0 + a0]; ++a0; } else { o = sI[b1 + a1]; ++a1; }
      idx[(size_t)(q0 + lane) * 8 + s] = o;
    }
  }
}

// ---------------------------------------------------------------------------
// Graph/elementwise glue kernels
// ---------------------------------------------------------------------------
__global__ void build_edges1(const float* __restrict__ pcd, const int* __restrict__ idx,
                             __bf16* __restrict__ e, int rows) {
  int row = blockIdx.x * blockDim.x + threadIdx.x;
  if (row >= rows) return;
  int i = row >> 4, j = idx[row];
  float xi0 = pcd[3 * i], xi1 = pcd[3 * i + 1], xi2 = pcd[3 * i + 2];
  __bf16* er = e + (size_t)row * 32;
  er[0] = (__bf16)xi0; er[1] = (__bf16)xi1; er[2] = (__bf16)xi2;
  er[3] = (__bf16)(pcd[3 * j] - xi0);
  er[4] = (__bf16)(pcd[3 * j + 1] - xi1);
  er[5] = (__bf16)(pcd[3 * j + 2] - xi2);
#pragma unroll
  for (int c = 6; c < 32; ++c) er[c] = (__bf16)0.0f;
}

__global__ void max_over_k(const float* __restrict__ z, float* __restrict__ out,
                           int N, int k, int C) {
  int i = blockIdx.x * blockDim.x + threadIdx.x;
  if (i >= N * C) return;
  int n = i / C, c = i % C;
  const float* base = z + (size_t)n * k * C + c;
  float m = base[0];
  for (int j = 1; j < k; ++j) m = fmaxf(m, base[(size_t)j * C]);
  out[i] = m;
}

__global__ void build_f128(const float* __restrict__ f64, __bf16* __restrict__ f128, int N) {
  int n = blockIdx.x * blockDim.x + threadIdx.x;
  if (n >= N) return;
  float mx = -3.4e38f;
#pragma unroll
  for (int c = 0; c < 64; ++c) mx = fmaxf(mx, f64[(size_t)n * 64 + c]);
#pragma unroll
  for (int c = 0; c < 64; ++c) f128[(size_t)n * 128 + c] = (__bf16)f64[(size_t)n * 64 + c];
#pragma unroll
  for (int c = 0; c < 64; ++c) f128[(size_t)n * 128 + 64 + c] = (__bf16)mx;
}

// fusion = [pcd;pcd]: both 2-NN of point i are its two copies -> pos_d == 0
__global__ void build_posd(const float* __restrict__ pcd, __bf16* __restrict__ pd, int rows) {
  int row = blockIdx.x * blockDim.x + threadIdx.x;
  if (row >= rows) return;
  int i = row >> 1;
  float d0 = pcd[3 * i] - pcd[3 * i];
  float d1 = pcd[3 * i + 1] - pcd[3 * i + 1];
  float d2 = pcd[3 * i + 2] - pcd[3 * i + 2];
  __bf16* r = pd + (size_t)row * 32;
  r[0] = (__bf16)d0; r[1] = (__bf16)d1; r[2] = (__bf16)d2;
#pragma unroll
  for (int c = 3; c < 32; ++c) r[c] = (__bf16)0.0f;
}

__global__ void build_ain(const float* __restrict__ feat, const float* __restrict__ delta,
                          __bf16* __restrict__ ain, int rows) {
  int i = blockIdx.x * blockDim.x + threadIdx.x;
  if (i >= rows * 32) return;
  int row = i >> 5, c = i & 31;
  int n = row >> 1;
  float fj = feat[(size_t)n * 32 + c];  // neighbor = the point's own copy
  ain[i] = (__bf16)((feat[(size_t)n * 32 + c] - fj) + delta[i]);
}

__global__ void softmax_combine(const float* __restrict__ att, const float* __restrict__ feat,
                                const float* __restrict__ delta, float* __restrict__ fct,
                                __bf16* __restrict__ fct_bf, int N) {
  int i = blockIdx.x * blockDim.x + threadIdx.x;
  if (i >= N * 32) return;
  int n = i >> 5, c = i & 31;
  float a0 = att[(size_t)(2 * n) * 32 + c];
  float a1 = att[(size_t)(2 * n + 1) * 32 + c];
  float m = fmaxf(a0, a1);
  float e0 = __expf(a0 - m), e1 = __expf(a1 - m);
  float w = 1.0f / (e0 + e1);
  float f = feat[i];
  float v0 = f + delta[(size_t)(2 * n) * 32 + c];
  float v1 = f + delta[(size_t)(2 * n + 1) * 32 + c];
  float o = (e0 * v0 + e1 * v1) * w;
  fct[i] = o;
  fct_bf[i] = (__bf16)o;
}

__global__ void rownorm32_bf(const float* __restrict__ f, float* __restrict__ nrm, int N) {
  int n = blockIdx.x * blockDim.x + threadIdx.x;
  if (n >= N) return;
  float s = 0.0f;
#pragma unroll
  for (int c = 0; c < 32; ++c) {
    float v = (float)(__bf16)f[(size_t)n * 32 + c];  // norm of the bf16-rounded row
    s += v * v;
  }
  nrm[n] = s;
}

__global__ void build_edges2(const float* __restrict__ f, const int* __restrict__ idx,
                             __bf16* __restrict__ e, int rows) {
  int row = blockIdx.x * blockDim.x + threadIdx.x;
  if (row >= rows) return;
  int i = row >> 3, j = idx[row];
  __bf16* er = e + (size_t)row * 64;
#pragma unroll
  for (int c = 0; c < 32; ++c) {
    float fi = f[(size_t)i * 32 + c];
    float fj = f[(size_t)j * 32 + c];
    er[c] = (__bf16)fi;
    er[32 + c] = (__bf16)(fj - fi);
  }
}

__global__ void f32_to_bf16(const float* __restrict__ x, __bf16* __restrict__ y, int n) {
  int i = blockIdx.x * blockDim.x + threadIdx.x;
  if (i < n) y[i] = (__bf16)x[i];
}

__global__ void final_out(const float* __restrict__ pcd, const float* __restrict__ d3,
                          float* __restrict__ out, int rows) {
  int r = blockIdx.x * blockDim.x + threadIdx.x;
  if (r >= rows) return;
  int i = r >> 1;
  out[3 * r + 0] = pcd[3 * i + 0] + d3[(size_t)r * 16 + 0];
  out[3 * r + 1] = pcd[3 * i + 1] + d3[(size_t)r * 16 + 1];
  out[3 * r + 2] = pcd[3 * i + 2] + d3[(size_t)r * 16 + 2];
}

// ---------------------------------------------------------------------------
// Host orchestration
// ---------------------------------------------------------------------------
extern "C" void kernel_launch(void* const* d_in, const int* in_sizes, int n_in,
                              void* d_out, int out_size, void* d_ws, size_t ws_size,
                              hipStream_t stream) {
  (void)n_in; (void)out_size; (void)ws_size;
  const float* pcd = (const float*)d_in[0];
  const float* P[41];
  for (int i = 0; i < 41; ++i) P[i] = (const float*)d_in[i];

  const int N  = in_sizes[0] / 3;  // 8192
  const int N2 = 2 * N;
  const int E1 = N * 16;
  const int E2 = N * 8;

  char* W = (char*)d_ws;
  size_t off = 0;
  auto alloc = [&](size_t bytes) -> void* {
    off = (off + 255) & ~(size_t)255;
    void* p = W + off;
    off += bytes;
    return p;
  };

  // bf16 transposed/padded weights
  __bf16* gcnW1T = (__bf16*)alloc(32 * 32 * 2);
  __bf16* gcnW2T = (__bf16*)alloc(64 * 32 * 2);
  __bf16* mlpW1T = (__bf16*)alloc(128 * 128 * 2);
  __bf16* mlpW2T = (__bf16*)alloc(32 * 128 * 2);
  __bf16* posW1T = (__bf16*)alloc(32 * 32 * 2);
  __bf16* posW2T = (__bf16*)alloc(32 * 32 * 2);
  __bf16* attW1T = (__bf16*)alloc(32 * 32 * 2);
  __bf16* attW2T = (__bf16*)alloc(32 * 32 * 2);
  __bf16* nsW1T  = (__bf16*)alloc(32 * 64 * 2);
  __bf16* nsW2T  = (__bf16*)alloc(64 * 32 * 2);
  __bf16* mdW1T  = (__bf16*)alloc(64 * 32 * 2);
  __bf16* mdW2T  = (__bf16*)alloc(64 * 64 * 2);
  __bf16* mdW3T  = (__bf16*)alloc(16 * 64 * 2);
  float*  b3pad  = (float*)alloc(16 * 4);
  float*  statsA = (float*)alloc(256 * 4);   // per-channel sum / sumsq
  float*  statsM = (float*)alloc(256 * 4);   // mean / rstd

  int*    idx1 = (int*)alloc((size_t)E1 * 4);
  __bf16* e1   = (__bf16*)alloc((size_t)E1 * 32 * 2);
  float*  z1   = (float*)alloc((size_t)E1 * 32 * 4);
  __bf16* h1   = (__bf16*)alloc((size_t)E1 * 32 * 2);
  float*  z2   = (float*)alloc((size_t)E1 * 64 * 4);
  float*  f64b = (float*)alloc((size_t)N * 64 * 4);
  __bf16* f128 = (__bf16*)alloc((size_t)N * 128 * 2);
  float*  mz1  = (float*)alloc((size_t)N * 128 * 4);
  __bf16* mh1  = (__bf16*)alloc((size_t)N * 128 * 2);
  float*  feat = (float*)alloc((size_t)N * 32 * 4);
  __bf16* posd = (__bf16*)alloc((size_t)N2 * 32 * 2);
  float*  pz1  = (float*)alloc((size_t)N2 * 32 * 4);
  __bf16* ph1  = (__bf16*)alloc((size_t)N2 * 32 * 2);
  float*  delt = (float*)alloc((size_t)N2 * 32 * 4);
  __bf16* ain  = (__bf16*)alloc((size_t)N2 * 32 * 2);
  float*  az1  = (float*)alloc((size_t)N2 * 32 * 4);
  __bf16* ah1  = (__bf16*)alloc((size_t)N2 * 32 * 2);
  float*  attv = (float*)alloc((size_t)N2 * 32 * 4);
  float*  fct  = (float*)alloc((size_t)N * 32 * 4);
  __bf16* fctb = (__bf16*)alloc((size_t)N * 32 * 2);
  float*  fnrm = (float*)alloc((size_t)N * 4);
  int*    idx2 = (int*)alloc((size_t)E2 * 4);
  __bf16* e2   = (__bf16*)alloc((size_t)E2 * 64 * 2);
  float*  nz1  = (float*)alloc((size_t)E2 * 32 * 4);
  __bf16* nh1  = (__bf16*)alloc((size_t)E2 * 32 * 2);
  float*  nz2  = (float*)alloc((size_t)E2 * 64 * 4);
  __bf16* fub  = (__bf16*)alloc((size_t)N2 * 32 * 2);
  float*  dz1  = (float*)alloc((size_t)N2 * 64 * 4);
  __bf16* dh1  = (__bf16*)alloc((size_t)N2 * 64 * 2);
  float*  dz2  = (float*)alloc((size_t)N2 * 64 * 4);
  __bf16* dh2  = (__bf16*)alloc((size_t)N2 * 64 * 2);
  float*  d3   = (float*)alloc((size_t)N2 * 16 * 4);

  float* out       = (float*)d_out;
  float* feat_out  = out + (size_t)N2 * 3;  // [N2,32] == [N,64] row-major view

  auto prep = [&](const float* w, __bf16* wT, int K, int Nn, int Kp, int Np) {
    prep_wT_kernel<<<cdiv(Np * Kp, 256), 256, 0, stream>>>(w, wT, K, Nn, Kp, Np);
  };
  auto stats_bn = [&](const float* z, __bf16* h, const float* g, const float* be,
                      int M, int Nc, int leaky) {
    zero_f32<<<1, 256, 0, stream>>>(statsA, 2 * Nc);
    colstats_accum<<<256, 256, 0, stream>>>(z, statsA, M, Nc);
    finalize_stats<<<1, 128, 0, stream>>>(statsA, statsM, M, Nc);
    bn_act_bf16<<<cdiv(M * Nc, 256), 256, 0, stream>>>(z, statsM, g, be, h, M, Nc, leaky);
  };

  // --- weight prep ---
  prep(P[1], gcnW1T, 6, 32, 32, 32);
  prep(P[5], gcnW2T, 32, 64, 32, 64);
  prep(P[7], mlpW1T, 128, 128, 128, 128);
  prep(P[11], mlpW2T, 128, 32, 128, 32);
  prep(P[13], posW1T, 3, 32, 32, 32);
  prep(P[17], posW2T, 32, 32, 32, 32);
  prep(P[19], attW1T, 32, 32, 32, 32);
  prep(P[23], attW2T, 32, 32, 32, 32);
  prep(P[25], nsW1T, 64, 32, 64, 32);
  prep(P[29], nsW2T, 32, 64, 32, 64);
  prep(P[31], mdW1T, 32, 64, 32, 64);
  prep(P[35], mdW2T, 64, 64, 64, 64);
  prep(P[39], mdW3T, 64, 3, 64, 16);
  pad_bias16<<<1, 16, 0, stream>>>(P[40], b3pad);

  // --- DynamicEdgeConv(3->64, k=16) ---
  knn_pos_kernel<<<N / 256, 256, 0, stream>>>(pcd, idx1, N);
  build_edges1<<<cdiv(E1, 256), 256, 0, stream>>>(pcd, idx1, e1, E1);
  gemm_bf16_wmma<2><<<dim3(E1 / 16, 1), 32, 0, stream>>>(e1, gcnW1T, P[2], z1, E1, 32, 32);
  stats_bn(z1, h1, P[3], P[4], E1, 32, /*leaky=*/1);
  gemm_bf16_wmma<4><<<dim3(E1 / 16, 1), 32, 0, stream>>>(h1, gcnW2T, P[6], z2, E1, 64, 32);
  max_over_k<<<cdiv(N * 64, 256), 256, 0, stream>>>(z2, f64b, N, 16, 64);

  // --- concat channel-max, MLP([128,128,32]) ---
  build_f128<<<cdiv(N, 256), 256, 0, stream>>>(f64b, f128, N);
  gemm_bf16_wmma<4><<<dim3(N / 16, 2), 32, 0, stream>>>(f128, mlpW1T, P[8], mz1, N, 128, 128);
  stats_bn(mz1, mh1, P[9], P[10], N, 128, /*leaky=*/0);
  gemm_bf16_wmma<2><<<dim3(N / 16, 1), 32, 0, stream>>>(mh1, mlpW2T, P[12], feat, N, 32, 128);

  // --- CrossTransformer (fusion = [pcd;pcd] -> neighbors are self-copies) ---
  build_posd<<<cdiv(N2, 256), 256, 0, stream>>>(pcd, posd, N2);
  gemm_bf16_wmma<2><<<dim3(N2 / 16, 1), 32, 0, stream>>>(posd, posW1T, P[14], pz1, N2, 32, 32);
  stats_bn(pz1, ph1, P[15], P[16], N2, 32, /*leaky=*/0);
  gemm_bf16_wmma<2><<<dim3(N2 / 16, 1), 32, 0, stream>>>(ph1, posW2T, P[18], delt, N2, 32, 32);
  build_ain<<<cdiv(N2 * 32, 256), 256, 0, stream>>>(feat, delt, ain, N2);
  gemm_bf16_wmma<2><<<dim3(N2 / 16, 1), 32, 0, stream>>>(ain, attW1T, P[20], az1, N2, 32, 32);
  stats_bn(az1, ah1, P[21], P[22], N2, 32, /*leaky=*/0);
  gemm_bf16_wmma<2><<<dim3(N2 / 16, 1), 32, 0, stream>>>(ah1, attW2T, P[24], attv, N2, 32, 32);
  softmax_combine<<<cdiv(N * 32, 256), 256, 0, stream>>>(attv, feat, delt, fct, fctb, N);

  // --- NodeShuffle (k=8 kNN in feature space via WMMA Gram tiles) ---
  rownorm32_bf<<<cdiv(N, 256), 256, 0, stream>>>(fct, fnrm, N);
  knn_feat_wmma<<<N / 64, 128, 0, stream>>>(fctb, fnrm, idx2, N);
  build_edges2<<<cdiv(E2, 256), 256, 0, stream>>>(fct, idx2, e2, E2);
  gemm_bf16_wmma<2><<<dim3(E2 / 16, 1), 32, 0, stream>>>(e2, nsW1T, P[26], nz1, E2, 32, 64);
  stats_bn(nz1, nh1, P[27], P[28], E2, 32, /*leaky=*/1);
  gemm_bf16_wmma<4><<<dim3(E2 / 16, 1), 32, 0, stream>>>(nh1, nsW2T, P[30], nz2, E2, 64, 32);
  // max over k=8 -> [N,64]; contiguous reinterpret as [2N,32] = returned feat
  max_over_k<<<cdiv(N * 64, 256), 256, 0, stream>>>(nz2, feat_out, N, 8, 64);

  // --- mlp_delta MLP([32,64,64,3]) on the upsampled features ---
  f32_to_bf16<<<cdiv(N2 * 32, 256), 256, 0, stream>>>(feat_out, fub, N2 * 32);
  gemm_bf16_wmma<4><<<dim3(N2 / 16, 1), 32, 0, stream>>>(fub, mdW1T, P[32], dz1, N2, 64, 32);
  stats_bn(dz1, dh1, P[33], P[34], N2, 64, /*leaky=*/0);
  gemm_bf16_wmma<4><<<dim3(N2 / 16, 1), 32, 0, stream>>>(dh1, mdW2T, P[36], dz2, N2, 64, 64);
  stats_bn(dz2, dh2, P[37], P[38], N2, 64, /*leaky=*/0);
  gemm_bf16_wmma<1><<<dim3(N2 / 16, 1), 32, 0, stream>>>(dh2, mdW3T, b3pad, d3, N2, 16, 64);

  final_out<<<cdiv(N2, 256), 256, 0, stream>>>(pcd, d3, out, N2);
}
